// Attention_83159156785571
// MI455X (gfx1250) — matrix operand order repaired and checked by reading, not compiled, not run on hardware
//
#include <hip/hip_runtime.h>
#include <hip/hip_bf16.h>
#include <math.h>

// ---------------------------------------------------------------------------
// Bahdanau additive attention, fused for MI455X (gfx1250).
//   B=32, T=2048, D=1024, U=1024
// Heavy op: tanh(values@W2 + q@W1 + b) @ Wv  -> 137 GFLOP GEMM, done with
// V_WMMA_F32_16X16X32_BF16 (fp32 accum), v_proj never materialized.
// ---------------------------------------------------------------------------

typedef __attribute__((ext_vector_type(16))) __bf16 v16bf;
typedef __attribute__((ext_vector_type(8)))  __bf16 v8bf;
typedef __attribute__((ext_vector_type(8)))  float  v8f;

#define BATCH 32
#define TLEN  2048
#define DDIM  1024
#define UDIM  1024
#define TM    64            // rows of `values` per workgroup in the GEMM kernel
#define LDA   1032          // padded bf16 LDS row stride (2064 B -> bank-conflict free)

// Branchless tanh: 1 - 2/(exp(2x)+1).  Lowers to v_exp_f32 + v_rcp_f32,
// no EXEC-mask divergence (unlike libm tanhf), saturates correctly at +/-inf.
__device__ __forceinline__ float tanh_fast(float x) {
    return 1.0f - 2.0f / (__expf(2.0f * x) + 1.0f);
}

// ---------------------------------------------------------------------------
// Kernel 1: W2 [D][U] fp32  ->  W2t [U][D] bf16 (column-contiguous for B frags)
// ---------------------------------------------------------------------------
__global__ __launch_bounds__(256)
void w2_prep_kernel(const float* __restrict__ W2, __bf16* __restrict__ W2t) {
    int idx = blockIdx.x * 256 + threadIdx.x;     // 1M elements
    int d = idx >> 10;
    int u = idx & 1023;
    W2t[(size_t)u * DDIM + d] = (__bf16)W2[idx];  // read coalesced in u
}

// ---------------------------------------------------------------------------
// Kernel 2: qpb[b][u] = query[b]·W1[:,u] + b1[u] + b2[u]   (tiny, fp32)
// (b2 folded in here so the GEMM epilogue needs a single bias load)
// ---------------------------------------------------------------------------
__global__ __launch_bounds__(256)
void qproj_kernel(const float* __restrict__ query, const float* __restrict__ W1,
                  const float* __restrict__ b1, const float* __restrict__ b2,
                  float* __restrict__ qpb) {
    int idx = blockIdx.x * 256 + threadIdx.x;     // 32768 outputs
    int b = idx >> 10;
    int u = idx & 1023;
    float acc = b1[u] + b2[u];
    const float* q = query + b * DDIM;
    #pragma unroll 4
    for (int d = 0; d < DDIM; ++d)
        acc = fmaf(q[d], W1[(size_t)d * UDIM + u], acc);  // coalesced in u
    qpb[idx] = acc;
}

// ---------------------------------------------------------------------------
// Kernel 3 (main): fused  score[g] = sum_u tanh(values[g]·W2[:,u] + b2[u] +
//                                              qp[b][u]) * Wv[u]
// One block = 64 rows (never crosses a batch boundary since 64 | 2048).
// 8 waves = 4 row-tiles x 2 column-groups. A staged in LDS as bf16.
// ---------------------------------------------------------------------------
extern __shared__ char cdna5_smem[];

__global__ __launch_bounds__(256)
void score_gemm_kernel(const float* __restrict__ values,   // [B*T, D]
                       const __bf16* __restrict__ W2t,     // [U, D] bf16
                       const float* __restrict__ qpb,      // [B, U] (incl b1+b2)
                       const float* __restrict__ Wv,       // [U]
                       float* __restrict__ scores)         // [B*T]
{
    __bf16* lds_a  = (__bf16*)cdna5_smem;                       // TM x LDA bf16
    float*  lds_sc = (float*)(cdna5_smem + TM * LDA * 2);       // TM floats

    const int tid  = threadIdx.x;
    const int wave = tid >> 5;
    const int lane = tid & 31;
    const long row_base = (long)blockIdx.x * TM;
    const int  b = (int)(row_base / TLEN);

    if (tid < TM) lds_sc[tid] = 0.0f;

    // ---- Stage A: 64 rows x 1024 fp32 -> bf16 LDS (padded rows) ----
    #pragma unroll
    for (int it = 0; it < (TM * DDIM) / (256 * 4); ++it) {
        int idx = (it * 256 + tid) * 4;
        int r = idx >> 10;
        int c = idx & 1023;
        const float4 v = *(const float4*)(values + (row_base + r) * (long)DDIM + c);
        union { __bf16 h[4]; uint2 u2; } pk;
        pk.h[0] = (__bf16)v.x; pk.h[1] = (__bf16)v.y;
        pk.h[2] = (__bf16)v.z; pk.h[3] = (__bf16)v.w;
        *(uint2*)&lds_a[r * LDA + c] = pk.u2;
    }
    __syncthreads();

    const int rt = wave >> 1;              // row tile 0..3
    const int cg = wave & 1;               // col group 0..1
    const int ln = lane & 15;
    const int kq = (lane & 16) ? 16 : 0;   // A-frag K contiguous; lanes 16..31 take K+16

    const __bf16* arow = lds_a + (rt * 16 + ln) * LDA + kq;

    float racc[8];
    #pragma unroll
    for (int j = 0; j < 8; ++j) racc[j] = 0.0f;

    // 32 column chunks of 16 per group (covers all 1024 U columns across 2 groups)
    for (int nc = 0; nc < 32; ++nc) {
        const int n0 = cg * 16 + nc * 32;
        // B fragment: col n0+ln, K contiguous; lanes 16..31 take K+16
        const __bf16* bcol = W2t + (size_t)(n0 + ln) * DDIM + ((lane & 16) ? 16 : 0);

        v8f acc = {};
        #pragma unroll 8
        for (int kk = 0; kk < DDIM / 32; ++kk) {
            const int k0 = kk * 32;
            v8bf alo = *(const v8bf*)(arow + k0);
            v8bf ahi = *(const v8bf*)(arow + k0 + 8);
            v16bf af = __builtin_shufflevector(alo, ahi,
                        0,1,2,3,4,5,6,7,8,9,10,11,12,13,14,15);
            v16bf bf = *(const v16bf*)(bcol + k0);
            acc = __builtin_amdgcn_wmma_f32_16x16x32_bf16(
                      false, af, false, bf, (short)0, acc, false, false);
        }

        // Fused epilogue: tanh(acc + qpb) * Wv, accumulate over N (branchless)
        const int n = n0 + ln;             // N index of this lane's acc elems
        const float addn = qpb[(size_t)b * UDIM + n];
        const float wvn  = Wv[n];
        #pragma unroll
        for (int j = 0; j < 8; ++j)
            racc[j] += tanh_fast(acc[j] + addn) * wvn;
    }

    // Reduce racc across each 16-lane half (N dim), then into per-row LDS slots
    #pragma unroll
    for (int j = 0; j < 8; ++j) {
        float t = racc[j];
        #pragma unroll
        for (int m = 1; m < 16; m <<= 1) t += __shfl_xor(t, m, 32);
        if (ln == 0) {
            // lane 0 -> rows rt*16+j ; lane 16 -> rows rt*16+8+j
            atomicAdd(&lds_sc[rt * 16 + j + ((lane & 16) ? 8 : 0)], t);
        }
    }
    __syncthreads();

    if (tid < TM) scores[row_base + tid] = lds_sc[tid];
}

// ---------------------------------------------------------------------------
// Kernel 4: softmax over T per batch -> attention weights (output #2)
// (bv shifts all scores equally -> softmax invariant -> omitted)
// ---------------------------------------------------------------------------
__global__ __launch_bounds__(256)
void softmax_kernel(const float* __restrict__ scores, float* __restrict__ weights) {
    __shared__ float red[256];
    const int b = blockIdx.x;
    const int tid = threadIdx.x;
    const float* s = scores + (size_t)b * TLEN;

    float mx = -INFINITY;
    for (int t = tid; t < TLEN; t += 256) mx = fmaxf(mx, s[t]);
    red[tid] = mx; __syncthreads();
    for (int off = 128; off > 0; off >>= 1) {
        if (tid < off) red[tid] = fmaxf(red[tid], red[tid + off]);
        __syncthreads();
    }
    mx = red[0]; __syncthreads();

    float sum = 0.0f;
    for (int t = tid; t < TLEN; t += 256) sum += __expf(s[t] - mx);
    red[tid] = sum; __syncthreads();
    for (int off = 128; off > 0; off >>= 1) {
        if (tid < off) red[tid] += red[tid + off];
        __syncthreads();
    }
    const float inv = 1.0f / red[0];

    for (int t = tid; t < TLEN; t += 256)
        weights[(size_t)b * TLEN + t] = __expf(s[t] - mx) * inv;
}

// ---------------------------------------------------------------------------
// Kernel 5: split-T weighted sum of values (coalesced float4), partials to ws
// ---------------------------------------------------------------------------
__global__ __launch_bounds__(256)
void context_partial_kernel(const float* __restrict__ values,
                            const float* __restrict__ weights,
                            float* __restrict__ partial) {
    const int b  = blockIdx.x >> 3;         // 32 batches
    const int ch = blockIdx.x & 7;          // 8 T-chunks of 256
    const int d4 = threadIdx.x;             // 256 threads x float4 = 1024 d
    const float* vb = values + ((size_t)b * TLEN + ch * 256) * DDIM + d4 * 4;
    const float* wb = weights + (size_t)b * TLEN + ch * 256;

    float4 acc = {0.f, 0.f, 0.f, 0.f};
    #pragma unroll 4
    for (int t = 0; t < 256; ++t) {
        const float w = wb[t];
        const float4 v = *(const float4*)(vb + (size_t)t * DDIM);
        acc.x = fmaf(w, v.x, acc.x); acc.y = fmaf(w, v.y, acc.y);
        acc.z = fmaf(w, v.z, acc.z); acc.w = fmaf(w, v.w, acc.w);
    }
    *(float4*)(partial + (size_t)blockIdx.x * DDIM + d4 * 4) = acc;
}

// ---------------------------------------------------------------------------
// Kernel 6: deterministic reduction of the 8 T-chunk partials -> context
// ---------------------------------------------------------------------------
__global__ __launch_bounds__(256)
void context_reduce_kernel(const float* __restrict__ partial, float* __restrict__ ctx) {
    const int idx = blockIdx.x * 256 + threadIdx.x;   // 32768
    const int b = idx >> 10;
    const int d = idx & 1023;
    float s = 0.0f;
    #pragma unroll
    for (int ch = 0; ch < 8; ++ch)
        s += partial[((size_t)(b * 8 + ch)) * DDIM + d];
    ctx[idx] = s;
}

// ---------------------------------------------------------------------------
// Host-side launch
// Inputs: 0=query[32,1024] 1=values[32,2048,1024] 2=W1[1024,1024] 3=b1[1024]
//         4=W2[1024,1024]  5=b2[1024] 6=Wv[1024,1] 7=bv[1] (unused:
//         constant shift is softmax-invariant and score is not an output)
// Output: context [32*1024] ++ attention_weights [32*2048]
// ---------------------------------------------------------------------------
extern "C" void kernel_launch(void* const* d_in, const int* in_sizes, int n_in,
                              void* d_out, int out_size, void* d_ws, size_t ws_size,
                              hipStream_t stream) {
    const float* query  = (const float*)d_in[0];
    const float* values = (const float*)d_in[1];
    const float* W1     = (const float*)d_in[2];
    const float* b1     = (const float*)d_in[3];
    const float* W2     = (const float*)d_in[4];
    const float* b2     = (const float*)d_in[5];
    const float* Wv     = (const float*)d_in[6];
    (void)in_sizes; (void)n_in; (void)ws_size; (void)out_size;

    float* out_ctx = (float*)d_out;                     // [32*1024]
    float* out_w   = (float*)d_out + BATCH * DDIM;      // [32*2048]

    // Workspace layout (all offsets 16B aligned):
    char* ws = (char*)d_ws;
    __bf16* W2t    = (__bf16*)ws;                               // 2 MB
    float*  qpb    = (float*)(ws + 2097152);                    // 128 KB
    float*  scores = (float*)(ws + 2097152 + 131072);           // 256 KB
    float*  part   = (float*)(ws + 2097152 + 131072 + 262144);  // 1 MB

    w2_prep_kernel<<<(DDIM * UDIM) / 256, 256, 0, stream>>>(W2, W2t);
    qproj_kernel<<<(BATCH * UDIM) / 256, 256, 0, stream>>>(query, W1, b1, b2, qpb);

    const size_t smem = (size_t)TM * LDA * sizeof(__bf16) + TM * sizeof(float);
    score_gemm_kernel<<<(BATCH * TLEN) / TM, 256, smem, stream>>>(
        values, W2t, qpb, Wv, scores);

    softmax_kernel<<<BATCH, 256, 0, stream>>>(scores, out_w);
    context_partial_kernel<<<BATCH * 8, 256, 0, stream>>>(values, out_w, part);
    context_reduce_kernel<<<(BATCH * DDIM) / 256, 256, 0, stream>>>(part, out_ctx);
}
